// WindowAttention_38929583571395
// MI455X (gfx1250) — matrix operand order repaired and checked by reading, not compiled, not run on hardware
//
#include <hip/hip_runtime.h>
#include <hip/hip_bf16.h>
#include <stdint.h>

// ---------------- problem constants ----------------
#define HEADS 6
#define NQ    343           // tokens per window (7*7*7)
#define NPAD  352           // 343 padded to multiple of 32 (22*16)
#define DIM   192
#define HD    32            // head dim
#define BATCH 256
#define NWIN  64
#define ROWS  (BATCH * NQ)  // 87808 (= 5488 * 16 exactly)
#define ROWT  (ROWS / 16)   // 5488

typedef __bf16 bf16_t;
typedef bf16_t v16bf __attribute__((ext_vector_type(16)));
typedef bf16_t v8bf  __attribute__((ext_vector_type(8)));
typedef float  v8f   __attribute__((ext_vector_type(8)));

static __device__ __forceinline__ bf16_t f2bf(float f) {
  unsigned u = __builtin_bit_cast(unsigned, f);
  unsigned r = u + 0x7fffu + ((u >> 16) & 1u);   // round-to-nearest-even
  return __builtin_bit_cast(bf16_t, (unsigned short)(r >> 16));
}

static __device__ __forceinline__ v8f wmma_bf16(v16bf a, v16bf b, v8f c) {
  // D = A(16x32) * B(32x16) + C(16x16), f32 accumulate
  return __builtin_amdgcn_wmma_f32_16x16x32_bf16(false, a, false, b, (short)0, c,
                                                 false, false);
}

// A fragment (16x32 bf16), src row-major [*, ld], rows row0..row0+15, K offset k0.
// Contiguous k order used here (same as load_b): lane holds row (lane&15); group g=lane>>4 holds K {g*16..g*16+15}.
static __device__ __forceinline__ v16bf load_a(const bf16_t* src, int row0, int ld, int k0) {
  int lane = threadIdx.x & 31;
  int g = lane >> 4, m = lane & 15;
  const bf16_t* p = src + (size_t)(row0 + m) * ld + k0 + g * 16;
  v8bf lo = *(const v8bf*)p;
  v8bf hi = *(const v8bf*)(p + 8);
  v16bf r;
#pragma unroll
  for (int i = 0; i < 8; ++i) { r[i] = lo[i]; r[i + 8] = hi[i]; }
  return r;
}

static __device__ __forceinline__ v16bf load_a_clamp(const bf16_t* src, int row0, int rmax,
                                                     int ld, int k0) {
  int lane = threadIdx.x & 31;
  int g = lane >> 4, m = lane & 15;
  int row = row0 + m; row = (row > rmax) ? rmax : row;
  const bf16_t* p = src + (size_t)row * ld + k0 + g * 16;
  v8bf lo = *(const v8bf*)p;
  v8bf hi = *(const v8bf*)(p + 8);
  v16bf r;
#pragma unroll
  for (int i = 0; i < 8; ++i) { r[i] = lo[i]; r[i + 8] = hi[i]; }
  return r;
}

// B fragment (32x16 bf16) from "NxK row-major" source (rows = N columns of B, K contiguous).
// Contiguous k order used here (same as load_a): lane holds N = lane&15; group g holds K {g*16 .. g*16+15} -> one 32B load.
static __device__ __forceinline__ v16bf load_b(const bf16_t* src, int n0, int ld, int k0) {
  int lane = threadIdx.x & 31;
  int g = lane >> 4, n = lane & 15;
  return *(const v16bf*)(src + (size_t)(n0 + n) * ld + k0 + g * 16);
}

// ---------------- stage 0: f32 -> bf16 convert (optionally scaled) ----------------
__global__ __launch_bounds__(256) void cvt_kernel(const float* __restrict__ src,
                                                  bf16_t* __restrict__ dst,
                                                  long n, float scale) {
  long i = (long)blockIdx.x * 256 + threadIdx.x;
  if (i < n) dst[i] = f2bf(src[i] * scale);
}

// ---------------- stage 1: relative-position bias table  bias[h][q][kpad] ----------------
__global__ __launch_bounds__(256) void bias_kernel(const float* __restrict__ rel_table,
                                                   float* __restrict__ bias_t) {
  long i = (long)blockIdx.x * 256 + threadIdx.x;
  const long total = (long)HEADS * NQ * NPAD;
  if (i >= total) return;
  int k = (int)(i % NPAD);
  long t = i / NPAD;
  int q = (int)(t % NQ);
  int h = (int)(t / NQ);
  float v = 0.0f;
  if (k < NQ) {
    int qd = q / 49, qh = (q / 7) % 7, qw = q % 7;
    int kd = k / 49, kh = (k / 7) % 7, kw = k % 7;
    int idx = ((qd - kd + 6) * 13 + (qh - kh + 6)) * 13 + (qw - kw + 6);
    v = rel_table[idx * HEADS + h];
  }
  bias_t[i] = v;
}

// ---------------- stage 2: QKV GEMM  (x[87808x192] @ W^T, 3 weights fused) ----------------
// one wave per (row-tile, n0): computes the q, k and v 16x16 tiles sharing A fragments.
__global__ __launch_bounds__(256) void qkv_kernel(const bf16_t* __restrict__ xb,
                                                  const bf16_t* __restrict__ wqb,
                                                  const bf16_t* __restrict__ wkb,
                                                  const bf16_t* __restrict__ wvb,
                                                  bf16_t* __restrict__ qb,
                                                  bf16_t* __restrict__ kb,
                                                  bf16_t* __restrict__ vt) {
  int wid = threadIdx.x >> 5;
  int lane = threadIdx.x & 31;
  int g = lane >> 4, ln = lane & 15;
  long wave = (long)blockIdx.x * 8 + wid;
  const long NT = (long)ROWT * 12;
  if (wave >= NT) return;
  int rt = (int)(wave / 12);
  int n0 = (int)(wave % 12) * 16;
  v8f accq = {}, acck = {}, accv = {};
#pragma unroll
  for (int k0 = 0; k0 < DIM; k0 += 32) {
    v16bf a = load_a(xb, rt * 16, DIM, k0);       // shared by all three GEMMs
    accq = wmma_bf16(a, load_b(wqb, n0, DIM, k0), accq);
    acck = wmma_bf16(a, load_b(wkb, n0, DIM, k0), acck);
    accv = wmma_bf16(a, load_b(wvb, n0, DIM, k0), accv);
  }
#pragma unroll
  for (int r = 0; r < 8; ++r) {
    int row = rt * 16 + r + 8 * g;       // < 87808 always (exact tiling)
    int b = row / NQ, n = row % NQ;
    int col = n0 + ln;
    int h = col >> 5, d = col & 31;
    size_t bh = (size_t)b * HEADS + h;
    qb[(bh * NQ   + n) * HD + d] = f2bf(accq[r]);   // [bh][n][32]
    kb[(bh * NPAD + n) * HD + d] = f2bf(acck[r]);   // [bh][352][32]
    vt[(bh * HD + d) * NPAD + n] = f2bf(accv[r]);   // [bh][32][352] transposed
  }
}

// ---------------- stage 3: flash attention ----------------
// block = 4 waves x 32; each wave owns TWO 16-query tiles (32 queries) and reuses the
// K/V B-fragments across both; block covers 128 queries of one (b,h); 3 q-blocks.
__global__ __launch_bounds__(128) void attn_kernel(const bf16_t* __restrict__ qb,
                                                   const bf16_t* __restrict__ kb,
                                                   const bf16_t* __restrict__ vt,
                                                   const float* __restrict__ bias_t,
                                                   const float* __restrict__ mask,
                                                   bf16_t* __restrict__ aob) {
  __shared__ __align__(64) bf16_t pstage[4][2][16 * 32];
  int wid  = threadIdx.x >> 5;
  int lane = threadIdx.x & 31;
  int g = lane >> 4, ln = lane & 15;
  int blk  = blockIdx.x;
  int qblk = blk % 3;                 // 3 blocks of 128 queries cover 343
  int bh   = blk / 3;                 // 0..1535
  int h = bh % HEADS, b = bh / HEADS;
  int q0 = qblk * 128 + wid * 32;     // wave's first query

  const bf16_t* Q  = qb + (size_t)bh * NQ * HD;
  const bf16_t* K  = kb + (size_t)bh * NPAD * HD;
  const bf16_t* V  = vt + (size_t)bh * HD * NPAD;
  const float* BIA = bias_t + (size_t)h * NQ * NPAD;
  const float* MSK = mask + (size_t)(b % NWIN) * NQ * NQ;

  v16bf qa[2];
  qa[0] = load_a_clamp(Q, q0,      NQ - 1, HD, 0);
  qa[1] = load_a_clamp(Q, q0 + 16, NQ - 1, HD, 0);

  float mrow[2][8], lrow[2][8];
  int qrow[2][8];
  v8f o0[2], o1[2];
#pragma unroll
  for (int t = 0; t < 2; ++t) {
    o0[t] = (v8f){}; o1[t] = (v8f){};
#pragma unroll
    for (int r = 0; r < 8; ++r) {
      mrow[t][r] = -1e30f; lrow[t][r] = 0.0f;
      int qq = q0 + t * 16 + r + 8 * g;
      qrow[t][r] = (qq > NQ - 1) ? (NQ - 1) : qq;
    }
  }

  for (int k0 = 0; k0 < NPAD; k0 += 32) {
    // S = Q * K^T : K fragments shared by both query tiles
    v16bf bk0 = load_b(K, k0,      HD, 0);
    v16bf bk1 = load_b(K, k0 + 16, HD, 0);
    v8f s0[2], s1[2];
#pragma unroll
    for (int t = 0; t < 2; ++t) {
      s0[t] = (v8f){}; s1[t] = (v8f){};
      s0[t] = wmma_bf16(qa[t], bk0, s0[t]);
      s1[t] = wmma_bf16(qa[t], bk1, s1[t]);
    }

    int key0 = k0 + ln;
    int key1 = k0 + 16 + ln;
    bool v0 = key0 < NQ, v1 = key1 < NQ;

#pragma unroll
    for (int t = 0; t < 2; ++t) {
#pragma unroll
      for (int r = 0; r < 8; ++r) {
        float a0 = v0 ? (BIA[(size_t)qrow[t][r] * NPAD + key0] +
                         MSK[(size_t)qrow[t][r] * NQ   + key0]) : 0.0f;
        float a1 = v1 ? (BIA[(size_t)qrow[t][r] * NPAD + key1] +
                         MSK[(size_t)qrow[t][r] * NQ   + key1]) : 0.0f;
        s0[t][r] = v0 ? (s0[t][r] + a0) : -1e30f;
        s1[t][r] = v1 ? (s1[t][r] + a1) : -1e30f;
      }
      // row max over 32 keys (rows live in one 16-lane half)
      float mstep[8];
#pragma unroll
      for (int r = 0; r < 8; ++r) mstep[r] = fmaxf(s0[t][r], s1[t][r]);
#pragma unroll
      for (int off = 1; off < 16; off <<= 1)
#pragma unroll
        for (int r = 0; r < 8; ++r) mstep[r] = fmaxf(mstep[r], __shfl_xor(mstep[r], off, 32));

      float fsc[8], rsum[8];
#pragma unroll
      for (int r = 0; r < 8; ++r) {
        float mn = fmaxf(mrow[t][r], mstep[r]);
        fsc[r] = __expf(mrow[t][r] - mn);
        mrow[t][r] = mn;
        s0[t][r] = __expf(s0[t][r] - mn);
        s1[t][r] = __expf(s1[t][r] - mn);
        rsum[r] = s0[t][r] + s1[t][r];
      }
#pragma unroll
      for (int off = 1; off < 16; off <<= 1)
#pragma unroll
        for (int r = 0; r < 8; ++r) rsum[r] += __shfl_xor(rsum[r], off, 32);
#pragma unroll
      for (int r = 0; r < 8; ++r) {
        lrow[t][r] = lrow[t][r] * fsc[r] + rsum[r];
        o0[t][r] *= fsc[r];
        o1[t][r] *= fsc[r];
      }
      // stage P_t (16x32) as bf16 in per-wave LDS
      bf16_t* ps = pstage[wid][t];
#pragma unroll
      for (int r = 0; r < 8; ++r) {
        int qq = r + 8 * g;
        ps[qq * 32 + ln]      = f2bf(s0[t][r]);
        ps[qq * 32 + 16 + ln] = f2bf(s1[t][r]);
      }
    }
    asm volatile("s_wait_dscnt 0x0" ::: "memory");   // wave-local LDS store->load ordering

    // O += P * V : V fragments shared by both query tiles
    v16bf bv0 = load_b(V, 0,  NPAD, k0);
    v16bf bv1 = load_b(V, 16, NPAD, k0);
#pragma unroll
    for (int t = 0; t < 2; ++t) {
      v16bf pa = load_a(pstage[wid][t], 0, 32, 0);
      o0[t] = wmma_bf16(pa, bv0, o0[t]);
      o1[t] = wmma_bf16(pa, bv1, o1[t]);
    }
  }
  // normalize and store [b][n][h*32+d] as bf16
#pragma unroll
  for (int t = 0; t < 2; ++t) {
#pragma unroll
    for (int r = 0; r < 8; ++r) {
      int qq = q0 + t * 16 + r + 8 * g;
      if (qq < NQ) {
        float inv = 1.0f / lrow[t][r];
        size_t base = ((size_t)b * NQ + qq) * DIM + h * HD;
        aob[base + ln]      = f2bf(o0[t][r] * inv);
        aob[base + 16 + ln] = f2bf(o1[t][r] * inv);
      }
    }
  }
}

// ---------------- stage 4: output projection + bias ----------------
// one wave per (row-tile, col-block of 4): A fragment reused across 4 WMMAs per k-step.
__global__ __launch_bounds__(256) void proj_kernel(const bf16_t* __restrict__ aob,
                                                   const bf16_t* __restrict__ pwb,
                                                   const float* __restrict__ pbias,
                                                   float* __restrict__ out) {
  int wid = threadIdx.x >> 5;
  int lane = threadIdx.x & 31;
  int g = lane >> 4, ln = lane & 15;
  long wave = (long)blockIdx.x * 8 + wid;
  const long NT = (long)ROWT * 3;
  if (wave >= NT) return;
  int rt = (int)(wave / 3);
  int n0 = (int)(wave % 3) * 64;
  v8f acc[4];
#pragma unroll
  for (int j = 0; j < 4; ++j) acc[j] = (v8f){};
#pragma unroll
  for (int k0 = 0; k0 < DIM; k0 += 32) {
    v16bf a = load_a(aob, rt * 16, DIM, k0);
#pragma unroll
    for (int j = 0; j < 4; ++j)
      acc[j] = wmma_bf16(a, load_b(pwb, n0 + j * 16, DIM, k0), acc[j]);
  }
#pragma unroll
  for (int j = 0; j < 4; ++j) {
#pragma unroll
    for (int r = 0; r < 8; ++r) {
      int row = rt * 16 + r + 8 * g;
      int col = n0 + j * 16 + ln;
      out[(size_t)row * DIM + col] = acc[j][r] + pbias[col];
    }
  }
}

// ---------------- launcher ----------------
extern "C" void kernel_launch(void* const* d_in, const int* in_sizes, int n_in,
                              void* d_out, int out_size, void* d_ws, size_t ws_size,
                              hipStream_t stream) {
  const float* x    = (const float*)d_in[0];
  const float* mask = (const float*)d_in[1];
  const float* wq   = (const float*)d_in[2];
  const float* wk   = (const float*)d_in[3];
  const float* wv   = (const float*)d_in[4];
  const float* rel  = (const float*)d_in[5];
  const float* pw   = (const float*)d_in[6];
  const float* pb   = (const float*)d_in[7];
  float* out = (float*)d_out;

  // workspace layout (~139.8 MB). xb and aob alias (disjoint live ranges).
  char* ws = (char*)d_ws;
  const size_t sz_xb   = (size_t)ROWS * DIM * 2;                 // 33,718,272
  const size_t sz_q    = (size_t)BATCH * HEADS * NQ * HD * 2;    // 33,718,272
  const size_t sz_kpad = (size_t)BATCH * HEADS * NPAD * HD * 2;  // 34,603,008
  const size_t sz_bias = (size_t)HEADS * NQ * NPAD * 4;          //  2,897,664
  const size_t sz_w    = (size_t)DIM * DIM * 2;                  //     73,728

  bf16_t* xb  = (bf16_t*)ws;                 // aliased with aob
  bf16_t* aob = (bf16_t*)ws;
  size_t off = sz_xb;
  bf16_t* qb  = (bf16_t*)(ws + off); off += sz_q;
  bf16_t* kb  = (bf16_t*)(ws + off); off += sz_kpad;
  bf16_t* vt  = (bf16_t*)(ws + off); off += sz_kpad;
  float*  bt  = (float*) (ws + off); off += sz_bias;
  bf16_t* wqb = (bf16_t*)(ws + off); off += sz_w;
  bf16_t* wkb = (bf16_t*)(ws + off); off += sz_w;
  bf16_t* wvb = (bf16_t*)(ws + off); off += sz_w;
  bf16_t* pwb = (bf16_t*)(ws + off); off += sz_w;

  // zero padded k/v buffers (pad rows/cols must read as 0)
  hipMemsetAsync(kb, 0, sz_kpad, stream);
  hipMemsetAsync(vt, 0, sz_kpad, stream);

  const float qscale = 0.17677669529663687f;   // 32^-0.5
  long nx = (long)ROWS * DIM;                  // 16,859,136 = 65,856 * 256
  cvt_kernel<<<65856, 256, 0, stream>>>(x, xb, nx, 1.0f);
  cvt_kernel<<<144, 256, 0, stream>>>(wq, wqb, (long)DIM * DIM, qscale);
  cvt_kernel<<<144, 256, 0, stream>>>(wk, wkb, (long)DIM * DIM, 1.0f);
  cvt_kernel<<<144, 256, 0, stream>>>(wv, wvb, (long)DIM * DIM, 1.0f);
  cvt_kernel<<<144, 256, 0, stream>>>(pw, pwb, (long)DIM * DIM, 1.0f);

  long nbias = (long)HEADS * NQ * NPAD;
  bias_kernel<<<(int)((nbias + 255) / 256), 256, 0, stream>>>(rel, bt);

  // QKV fused GEMM: 5488 row tiles * 12 col tiles = 65,856 waves / 8 per block
  qkv_kernel<<<8232, 256, 0, stream>>>(xb, wqb, wkb, wvb, qb, kb, vt);

  // attention: (256 b * 6 h * 3 qblocks) blocks of 4 waves (32 queries/wave)
  attn_kernel<<<4608, 128, 0, stream>>>(qb, kb, vt, bt, mask, aob);

  // projection: 5488 row tiles * 3 col-blocks = 16,464 waves / 8 per block
  proj_kernel<<<2058, 256, 0, stream>>>(aob, pwb, pb, out);
}
